// VanillaRNN_13288628814136
// MI455X (gfx1250) — matrix operand order repaired and checked by reading, not compiled, not run on hardware
//
#include <hip/hip_runtime.h>
#include <hip/hip_bf16.h>
#include <math.h>

// ---------------------------------------------------------------------------
// VanillaRNN on MI455X (gfx1250), bf16 WMMA w/ fp32 accumulation.
//   h_{t+1} = tanh([h_t | x_t] @ Wcomb + bh),  Wcomb = [Whh ; Wxh]  (K=1152)
//   out     = h_T @ Whp + bp
// Persistent step kernel (32 blocks) with device-scope barrier between steps.
// Workspace usage (~37.3 MB): x_bf16 | WcombT | WhpT | h_a | h_b | counters
// ---------------------------------------------------------------------------

#define BATCH 256
#define SEQ   512
#define DIN   128
#define DH    1024
#define DC    128
#define KTOT  (DH + DIN)   // 1152

typedef __attribute__((ext_vector_type(16))) __bf16 v16bf;
typedef __attribute__((ext_vector_type(8)))  __bf16 v8bf;
typedef __attribute__((ext_vector_type(8)))  float  v8f;

union FragU { v16bf v; v8bf h[2]; };

// ---------------- one-time packing kernels ----------------

__global__ void k_convert_x(const float* __restrict__ x, __bf16* __restrict__ xb, int n) {
    int i = blockIdx.x * blockDim.x + threadIdx.x;
    if (i < n) xb[i] = (__bf16)x[i];
}

// WcombT[n][k] = (k < DH) ? Whh[k][n] : Wxh[k-DH][n]   (bf16, row-major [DH][KTOT])
__global__ void k_pack_wcomb(const float* __restrict__ Whh, const float* __restrict__ Wxh,
                             __bf16* __restrict__ wc) {
    int idx = blockIdx.x * blockDim.x + threadIdx.x;
    if (idx >= DH * KTOT) return;
    int n = idx / KTOT;
    int k = idx % KTOT;
    float v = (k < DH) ? Whh[(size_t)k * DH + n] : Wxh[(size_t)(k - DH) * DH + n];
    wc[idx] = (__bf16)v;
}

// WhpT[n][k] = Whp[k][n]   (bf16, row-major [DC][DH])
__global__ void k_pack_whp(const float* __restrict__ Whp, __bf16* __restrict__ wp) {
    int idx = blockIdx.x * blockDim.x + threadIdx.x;
    if (idx >= DC * DH) return;
    int n = idx / DH;
    int k = idx % DH;
    wp[idx] = (__bf16)Whp[(size_t)k * DC + n];
}

// zero h0 and the per-step barrier counters (replay-deterministic)
__global__ void k_init(unsigned* __restrict__ h_a_u32, unsigned* __restrict__ cnt) {
    int idx = blockIdx.x * blockDim.x + threadIdx.x;
    if (idx < (BATCH * DH * 2) / 4) h_a_u32[idx] = 0u;   // 512KB of bf16 zeros
    if (idx < SEQ) cnt[idx] = 0u;
}

// ---------------- persistent recurrence kernel ----------------
// grid = 32 blocks x 256 threads (8 waves). Block covers 32(M) x 256(N);
// wave covers 16(M) x 64(N) -> 4 f32 16x16 accumulators.

__global__ void __launch_bounds__(256) k_rnn_steps(
    const __bf16* __restrict__ xb,      // [BATCH, SEQ, DIN] bf16
    const __bf16* __restrict__ wcombT,  // [DH, KTOT] bf16
    const float*  __restrict__ bh,      // [DH] f32
    __bf16* __restrict__ h_a,           // [BATCH, DH] bf16 (ping)
    __bf16* __restrict__ h_b,           // [BATCH, DH] bf16 (pong)
    unsigned* __restrict__ cnt)         // [SEQ] barrier counters
{
    const int tid  = threadIdx.x;
    const int lane = tid & 31;
    const int w    = tid >> 5;
    const int half = lane >> 4;   // wave32 half (VGPR layout split)
    const int l16  = lane & 15;

    const int m0 = (blockIdx.x & 7) * 32 + (w & 1) * 16;   // 8 M-blocks of 32
    const int n0 = (blockIdx.x >> 3) * 256 + (w >> 1) * 64; // 4 N-blocks of 256

    const int arow = m0 + l16;           // A-matrix row handled by this lane

    float bhv[4];
    #pragma unroll
    for (int j = 0; j < 4; ++j) bhv[j] = bh[n0 + j * 16 + l16];

    const unsigned nblocks = gridDim.x;

    for (int t = 0; t < SEQ; ++t) {
        const __bf16* hin  = (t & 1) ? h_b : h_a;
        __bf16*       hout = (t & 1) ? h_a : h_b;

        v8f acc[4];
        #pragma unroll
        for (int j = 0; j < 4; ++j) {
            #pragma unroll
            for (int r = 0; r < 8; ++r) acc[j][r] = bhv[j];
        }

        // ---- K = 0..1023 : recurrent term h @ Whh ----
        const __bf16* aH = hin + (size_t)arow * DH + half * 16;
        #pragma unroll 4
        for (int k0 = 0; k0 < DH; k0 += 32) {
            FragU a;
            a.h[0] = *(const v8bf*)(aH + k0);         // K = k0 + half*16 .. +7
            a.h[1] = *(const v8bf*)(aH + k0 + 8);     // K = k0 + half*16+8 .. +15
            #pragma unroll
            for (int j = 0; j < 4; ++j) {
                const __bf16* bptr =
                    wcombT + (size_t)(n0 + j * 16 + l16) * KTOT + k0 + half * 16;
                FragU b;
                b.h[0] = *(const v8bf*)(bptr);        // K = k0 + half*16 .. +7
                b.h[1] = *(const v8bf*)(bptr + 8);    // K = k0 + half*16+8 .. +15
                acc[j] = __builtin_amdgcn_wmma_f32_16x16x32_bf16(
                    false, a.v, false, b.v, (short)0, acc[j], false, false);
            }
        }

        // ---- K = 1024..1151 : input term x_t @ Wxh (fused) ----
        const __bf16* aX = xb + ((size_t)arow * SEQ + t) * DIN + half * 16;
        #pragma unroll
        for (int k0 = 0; k0 < DIN; k0 += 32) {
            FragU a;
            a.h[0] = *(const v8bf*)(aX + k0);
            a.h[1] = *(const v8bf*)(aX + k0 + 8);
            #pragma unroll
            for (int j = 0; j < 4; ++j) {
                const __bf16* bptr =
                    wcombT + (size_t)(n0 + j * 16 + l16) * KTOT + DH + k0 + half * 16;
                FragU b;
                b.h[0] = *(const v8bf*)(bptr);
                b.h[1] = *(const v8bf*)(bptr + 8);
                acc[j] = __builtin_amdgcn_wmma_f32_16x16x32_bf16(
                    false, a.v, false, b.v, (short)0, acc[j], false, false);
            }
        }

        // ---- epilogue: tanh, store bf16 h_{t+1} ----
        #pragma unroll
        for (int j = 0; j < 4; ++j) {
            const int col = n0 + j * 16 + l16;
            #pragma unroll
            for (int r = 0; r < 8; ++r) {
                const int row = m0 + half * 8 + r;    // C/D layout: M = r + 8*half
                hout[(size_t)row * DH + col] = (__bf16)tanhf(acc[j][r]);
            }
        }

        // ---- device-scope barrier between steps (per-step counter) ----
        if (t != SEQ - 1) {
            __syncthreads();
            if (tid == 0) {
                __hip_atomic_fetch_add(&cnt[t], 1u, __ATOMIC_RELEASE,
                                       __HIP_MEMORY_SCOPE_AGENT);
                while (__hip_atomic_load(&cnt[t], __ATOMIC_ACQUIRE,
                                         __HIP_MEMORY_SCOPE_AGENT) < nblocks) {
                    __builtin_amdgcn_s_sleep(1);
                }
            }
            __syncthreads();
        }
    }
}

// ---------------- final projection: out = h_T @ Whp + bp ----------------
// grid = 4 blocks x 256 threads; wave covers 16(M) x 64(N), K = 1024.

__global__ void __launch_bounds__(256) k_final(
    const __bf16* __restrict__ hfin,   // [BATCH, DH] bf16 (ends in h_a: t=511 odd writes h_a)
    const __bf16* __restrict__ whpT,   // [DC, DH] bf16
    const float*  __restrict__ bpv_in, // [DC] f32
    float* __restrict__ out)           // [BATCH, DC] f32
{
    const int tid  = threadIdx.x;
    const int lane = tid & 31;
    const int w    = tid >> 5;
    const int half = lane >> 4;
    const int l16  = lane & 15;

    const int m0 = blockIdx.x * 64 + (w & 3) * 16;
    const int n0 = (w >> 2) * 64;
    const int arow = m0 + l16;

    v8f acc[4];
    #pragma unroll
    for (int j = 0; j < 4; ++j) {
        float b = bpv_in[n0 + j * 16 + l16];
        #pragma unroll
        for (int r = 0; r < 8; ++r) acc[j][r] = b;
    }

    const __bf16* aP = hfin + (size_t)arow * DH + half * 16;
    #pragma unroll 4
    for (int k0 = 0; k0 < DH; k0 += 32) {
        FragU a;
        a.h[0] = *(const v8bf*)(aP + k0);
        a.h[1] = *(const v8bf*)(aP + k0 + 8);
        #pragma unroll
        for (int j = 0; j < 4; ++j) {
            const __bf16* bptr = whpT + (size_t)(n0 + j * 16 + l16) * DH + k0 + half * 16;
            FragU b;
            b.h[0] = *(const v8bf*)(bptr);
            b.h[1] = *(const v8bf*)(bptr + 8);
            acc[j] = __builtin_amdgcn_wmma_f32_16x16x32_bf16(
                false, a.v, false, b.v, (short)0, acc[j], false, false);
        }
    }

    #pragma unroll
    for (int j = 0; j < 4; ++j) {
        const int col = n0 + j * 16 + l16;
        #pragma unroll
        for (int r = 0; r < 8; ++r) {
            const int row = m0 + half * 8 + r;
            out[(size_t)row * DC + col] = acc[j][r];
        }
    }
}

// ---------------- host launch ----------------

extern "C" void kernel_launch(void* const* d_in, const int* in_sizes, int n_in,
                              void* d_out, int out_size, void* d_ws, size_t ws_size,
                              hipStream_t stream) {
    const float* x   = (const float*)d_in[0];   // [256,512,128]
    const float* Wxh = (const float*)d_in[1];   // [128,1024]
    const float* Whh = (const float*)d_in[2];   // [1024,1024]
    const float* Whp = (const float*)d_in[3];   // [1024,128]
    const float* bh  = (const float*)d_in[4];   // [1024]
    const float* bp  = (const float*)d_in[5];   // [128]
    float* out = (float*)d_out;                 // [256,128]

    char* ws = (char*)d_ws;
    // workspace layout (all offsets 256B-aligned), total ~37.3 MB
    __bf16*   xb  = (__bf16*)(ws + 0);              // 33,554,432 B
    __bf16*   wc  = (__bf16*)(ws + 33554432);       //  2,359,296 B
    __bf16*   wp  = (__bf16*)(ws + 35913728);       //    262,144 B
    __bf16*   h_a = (__bf16*)(ws + 36175872);       //    524,288 B
    __bf16*   h_b = (__bf16*)(ws + 36700160);       //    524,288 B
    unsigned* cnt = (unsigned*)(ws + 37224448);     //      2,048 B

    const int nx = BATCH * SEQ * DIN;               // 16,777,216
    k_convert_x<<<(nx + 255) / 256, 256, 0, stream>>>(x, xb, nx);
    k_pack_wcomb<<<(DH * KTOT + 255) / 256, 256, 0, stream>>>(Whh, Wxh, wc);
    k_pack_whp<<<(DC * DH + 255) / 256, 256, 0, stream>>>(Whp, wp);
    k_init<<<512, 256, 0, stream>>>((unsigned*)h_a, cnt);

    k_rnn_steps<<<32, 256, 0, stream>>>(xb, wc, bh, h_a, h_b, cnt);

    // SEQ=512 steps: last step (t=511, odd) writes h_a -> final hidden state
    k_final<<<4, 256, 0, stream>>>(h_a, wp, bp, out);
}
